// InferDecoder_55155970015698
// MI455X (gfx1250) — compile-verified, matrix operand order checked by reading, not run on hardware
//
#include <hip/hip_runtime.h>
#include <math.h>

// ---------------------------------------------------------------------------
// LSTM decoder inference for MI455X (gfx1250, wave32, WMMA).
// N=32, T=64, V=32000, E=512, H=1024.
//
// - Weights converted once per launch to bf16 in WMMA B-fragment order
//   (~77MB, stays resident in the 192MB L2 across all 64 steps).
// - Activations (h, emb[tok]) are packed once per step into WMMA A-fragment
//   bf16 order, so GEMM inner loops are pure {b128 loads -> v_wmma}.
// - Each wave processes 2 N-tiles x 2 M-tiles (4 accumulators) to amortize
//   fragment loads: 4 WMMAs per 8 b128 loads.
// ---------------------------------------------------------------------------

typedef __bf16 bf16_t;
typedef __attribute__((ext_vector_type(16))) __bf16 v16bf;
typedef __attribute__((ext_vector_type(8)))  float  v8f;

#define N_BATCH 32
#define T_STEPS 64
#define VOCAB   32000
#define EMB     512
#define HID     1024
#define G4      (4 * HID)
#define START_TOK 1

// ---------------------------------------------------------------------------
// Fragment index helpers (ISA 7.12.2, 16-bit matrices, wave32):
//   A 16x32: lane holds row M = lane%16; elems 0..7 = K[8*(lane/16)+0..7],
//            elems 8..15 = K[16+8*(lane/16)+0..7] within the 32-K chunk.
//   B 32x16: lane holds col N = lane%16; same K split across lane halves.
// Packed storage: one 512-element block per (tile, kchunk); lane's 16
// elements contiguous -> a v16bf fragment is one 32-byte coalesced load.
// ---------------------------------------------------------------------------
__device__ __host__ inline int frag_k_of(int lane, int j) {
    int khalf = lane >> 4;
    return (j < 8) ? (khalf * 8 + j) : (16 + khalf * 8 + (j - 8));
}

// Inverse: k-within-chunk (0..31) -> (lane-half, elem index)
__device__ inline void frag_pos_of(int ko, int& khalf, int& jj) {
    int q = ko >> 3;               // which 8-group: 0..3
    khalf = q & 1;
    jj = (ko & 7) + ((q >> 1) << 3);
}

// ---------------------------------------------------------------------------
// Pack f32 row-major (K x Ncols) weights into bf16 WMMA-B fragment order.
// ---------------------------------------------------------------------------
__global__ void pack_bf16_kernel(const float* __restrict__ src,
                                 bf16_t* __restrict__ dst,
                                 int K, int Ncols) {
    long long tid = (long long)blockIdx.x * blockDim.x + threadIdx.x;
    long long total = (long long)K * Ncols;
    if (tid >= total) return;
    int blk  = (int)(tid >> 9);
    int r    = (int)(tid & 511);
    int lane = r >> 4;
    int j    = r & 15;
    int kChunks = K >> 5;
    int nt = blk / kChunks;
    int kc = blk - nt * kChunks;
    int k = kc * 32 + frag_k_of(lane, j);
    int n = nt * 16 + (lane & 15);
    dst[tid] = (bf16_t)src[(long long)k * Ncols + n];
}

// ---------------------------------------------------------------------------
// Pack a (32 x K) f32 activation matrix into bf16 A-fragment order
// (2 M-tiles x K/32 chunks x 512).  Used once for h0 at init.
// ---------------------------------------------------------------------------
__global__ void pack_act_kernel(const float* __restrict__ src,
                                bf16_t* __restrict__ dst, int K) {
    int tid = blockIdx.x * blockDim.x + threadIdx.x;
    int kChunks = K >> 5;
    int total = 2 * kChunks * 512;
    if (tid >= total) return;
    int blk  = tid >> 9;
    int r    = tid & 511;
    int lane = r >> 4;
    int j    = r & 15;
    int m  = blk / kChunks;
    int kc = blk - m * kChunks;
    int k   = kc * 32 + frag_k_of(lane, j);
    int row = m * 16 + (lane & 15);
    dst[tid] = (bf16_t)src[row * K + k];
}

// ---------------------------------------------------------------------------
// Per-step: gather emb[tok] rows straight into bf16 A-fragment order.
// ---------------------------------------------------------------------------
__global__ void pack_emb_kernel(const float* __restrict__ emb,
                                const int* __restrict__ tok,
                                bf16_t* __restrict__ dst) {
    int tid = blockIdx.x * blockDim.x + threadIdx.x;   // 32*512 threads
    int blk  = tid >> 9;
    int r    = tid & 511;
    int lane = r >> 4;
    int j    = r & 15;
    int m  = blk / (EMB / 32);
    int kc = blk - m * (EMB / 32);
    int k   = kc * 32 + frag_k_of(lane, j);
    int row = m * 16 + (lane & 15);
    dst[tid] = (bf16_t)emb[(size_t)tok[row] * EMB + k];
}

// ---------------------------------------------------------------------------
// GEMM core: C(32 x 16*2) += Apacked(32 x K) * Bpacked(K x 16*2)
// One wave; nt0 = first of two adjacent N-tiles.
// ---------------------------------------------------------------------------
__device__ inline void wmma_mac_2x2(const bf16_t* __restrict__ aP,
                                    const bf16_t* __restrict__ bP,
                                    int nt0, int kChunks, int lane,
                                    v8f& acc00, v8f& acc10,
                                    v8f& acc01, v8f& acc11) {
    const bf16_t* ap0 = aP + (size_t)lane * 16;
    const bf16_t* ap1 = ap0 + (size_t)kChunks * 512;
    const bf16_t* bp0 = bP + (size_t)nt0 * kChunks * 512 + lane * 16;
    const bf16_t* bp1 = bp0 + (size_t)kChunks * 512;
    for (int kc = 0; kc < kChunks; ++kc) {
        v16bf a0 = *(const v16bf*)(ap0 + (size_t)kc * 512);
        v16bf a1 = *(const v16bf*)(ap1 + (size_t)kc * 512);
        v16bf b0 = *(const v16bf*)(bp0 + (size_t)kc * 512);
        v16bf b1 = *(const v16bf*)(bp1 + (size_t)kc * 512);
        acc00 = __builtin_amdgcn_wmma_f32_16x16x32_bf16(false, a0, false, b0,
                                                        (short)0, acc00, false, false);
        acc10 = __builtin_amdgcn_wmma_f32_16x16x32_bf16(false, a1, false, b0,
                                                        (short)0, acc10, false, false);
        acc01 = __builtin_amdgcn_wmma_f32_16x16x32_bf16(false, a0, false, b1,
                                                        (short)0, acc01, false, false);
        acc11 = __builtin_amdgcn_wmma_f32_16x16x32_bf16(false, a1, false, b1,
                                                        (short)0, acc11, false, false);
    }
}

__device__ inline void store_acc(float* __restrict__ out, int ldc, int ntile,
                                 int lane, float bias, const v8f& accLo,
                                 const v8f& accHi) {
    int khalf = lane >> 4;
    int col   = ntile * 16 + (lane & 15);
#pragma unroll
    for (int j = 0; j < 8; ++j) {
        int row = khalf * 8 + j;   // C layout: VGPR j -> M=j (+8 for hi lanes)
        out[(size_t)row        * ldc + col] = accLo[j] + bias;
        out[(size_t)(16 + row) * ldc + col] = accHi[j] + bias;
    }
}

// ---------------------------------------------------------------------------
// gates = e @ Wk + h @ Wr + b_lstm   (32 x 4096)
// 4 waves/block, 2 N-tiles/wave -> 32 blocks cover 256 N-tiles.
// ---------------------------------------------------------------------------
__global__ __launch_bounds__(128) void gates_kernel(
    const bf16_t* __restrict__ eP, const bf16_t* __restrict__ WkP,
    const bf16_t* __restrict__ hA, const bf16_t* __restrict__ WrP,
    const float* __restrict__ b_lstm, float* __restrict__ gates) {
    int wave = threadIdx.x >> 5;
    int lane = threadIdx.x & 31;
    int nt0  = (blockIdx.x * 4 + wave) * 2;

    v8f acc00 = {}, acc10 = {}, acc01 = {}, acc11 = {};
    wmma_mac_2x2(eP, WkP, nt0, EMB / 32, lane, acc00, acc10, acc01, acc11);
    wmma_mac_2x2(hA, WrP, nt0, HID / 32, lane, acc00, acc10, acc01, acc11);

    float bias0 = b_lstm[nt0 * 16 + (lane & 15)];
    float bias1 = b_lstm[(nt0 + 1) * 16 + (lane & 15)];
    store_acc(gates, G4, nt0,     lane, bias0, acc00, acc10);
    store_acc(gates, G4, nt0 + 1, lane, bias1, acc01, acc11);
}

// ---------------------------------------------------------------------------
// LSTM pointwise; writes c (f32) and h directly in packed bf16 A-fragment
// order (the only form downstream consumers need).
// ---------------------------------------------------------------------------
__global__ __launch_bounds__(256) void lstm_kernel(
    const float* __restrict__ gates, float* __restrict__ c,
    bf16_t* __restrict__ hA) {
    int tid = blockIdx.x * blockDim.x + threadIdx.x;
    if (tid >= N_BATCH * HID) return;
    int n = tid / HID;
    int k = tid - n * HID;
    const float* g = gates + (size_t)n * G4;
    float gi = g[k];
    float gf = g[k + HID];
    float gg = g[k + 2 * HID];
    float go = g[k + 3 * HID];
    gi = 1.f / (1.f + __expf(-gi));
    gf = 1.f / (1.f + __expf(-gf));
    go = 1.f / (1.f + __expf(-go));
    gg = tanhf(gg);
    float cn = gf * c[tid] + gi * gg;
    c[tid] = cn;
    float hv = go * tanhf(cn);

    // scatter into A-fragment layout
    int m = n >> 4, rowin = n & 15;
    int kc = k >> 5, ko = k & 31;
    int khalf, jj;
    frag_pos_of(ko, khalf, jj);
    int lane = khalf * 16 + rowin;
    hA[((size_t)(m * (HID / 32) + kc)) * 512 + lane * 16 + jj] = (bf16_t)hv;
}

// ---------------------------------------------------------------------------
// logits = h @ Wfc + bfc   (32 x 32000); 2000 N-tiles -> 250 blocks.
// ---------------------------------------------------------------------------
__global__ __launch_bounds__(128) void logits_kernel(
    const bf16_t* __restrict__ hA, const bf16_t* __restrict__ WfcP,
    const float* __restrict__ bfc, float* __restrict__ logits) {
    int wave = threadIdx.x >> 5;
    int lane = threadIdx.x & 31;
    int nt0  = (blockIdx.x * 4 + wave) * 2;

    v8f acc00 = {}, acc10 = {}, acc01 = {}, acc11 = {};
    wmma_mac_2x2(hA, WfcP, nt0, HID / 32, lane, acc00, acc10, acc01, acc11);

    float bias0 = bfc[nt0 * 16 + (lane & 15)];
    float bias1 = bfc[(nt0 + 1) * 16 + (lane & 15)];
    store_acc(logits, VOCAB, nt0,     lane, bias0, acc00, acc10);
    store_acc(logits, VOCAB, nt0 + 1, lane, bias1, acc01, acc11);
}

// ---------------------------------------------------------------------------
// Softmax + argmax per batch row; writes probs slice of d_out, next token
// into tok state, and token value into the d_out tail.
// ---------------------------------------------------------------------------
__global__ __launch_bounds__(256) void softmax_kernel(
    const float* __restrict__ logits, float* __restrict__ probs_out,
    float* __restrict__ tok_out, int* __restrict__ tok, int t) {
    __shared__ float sv[256];
    __shared__ int   si[256];
    int n = blockIdx.x;
    int tid = threadIdx.x;
    const float* row = logits + (size_t)n * VOCAB;

    float best = -INFINITY;
    int   bi   = 0;
    for (int v = tid; v < VOCAB; v += 256) {
        float x = row[v];
        if (x > best) { best = x; bi = v; }
    }
    sv[tid] = best; si[tid] = bi;
    __syncthreads();
    for (int s = 128; s > 0; s >>= 1) {
        if (tid < s) {
            if (sv[tid + s] > sv[tid] ||
                (sv[tid + s] == sv[tid] && si[tid + s] < si[tid])) {
                sv[tid] = sv[tid + s]; si[tid] = si[tid + s];
            }
        }
        __syncthreads();
    }
    float mx = sv[0];
    int amax = si[0];
    __syncthreads();

    float sum = 0.f;
    for (int v = tid; v < VOCAB; v += 256) sum += __expf(row[v] - mx);
    sv[tid] = sum;
    __syncthreads();
    for (int s = 128; s > 0; s >>= 1) {
        if (tid < s) sv[tid] += sv[tid + s];
        __syncthreads();
    }
    float inv = 1.f / sv[0];

    float* out = probs_out + ((size_t)n * T_STEPS + t) * VOCAB;
    for (int v = tid; v < VOCAB; v += 256) out[v] = __expf(row[v] - mx) * inv;

    if (tid == 0) {
        tok[n] = amax;
        tok_out[(size_t)n * T_STEPS + t] = (float)amax;
    }
}

// ---------------------------------------------------------------------------
// Init: c <- c0, tok <- START
// ---------------------------------------------------------------------------
__global__ void init_kernel(const float* __restrict__ c0,
                            float* __restrict__ c, int* __restrict__ tok) {
    int tid = blockIdx.x * blockDim.x + threadIdx.x;
    if (tid < N_BATCH * HID) c[tid] = c0[tid];
    if (tid < N_BATCH) tok[tid] = START_TOK;
}

// ---------------------------------------------------------------------------

static inline size_t align256(size_t x) { return (x + 255) & ~(size_t)255; }

extern "C" void kernel_launch(void* const* d_in, const int* in_sizes, int n_in,
                              void* d_out, int out_size, void* d_ws, size_t ws_size,
                              hipStream_t stream) {
    (void)in_sizes; (void)n_in; (void)out_size; (void)ws_size;
    const float* h0     = (const float*)d_in[0];
    const float* c0     = (const float*)d_in[1];
    const float* emb    = (const float*)d_in[2];
    const float* Wk     = (const float*)d_in[3];
    const float* Wr     = (const float*)d_in[4];
    const float* b_lstm = (const float*)d_in[5];
    const float* Wfc    = (const float*)d_in[6];
    const float* bfc    = (const float*)d_in[7];

    char* ws = (char*)d_ws;
    size_t off = 0;
    bf16_t* WkP  = (bf16_t*)(ws + off); off = align256(off + (size_t)EMB * G4 * 2);
    bf16_t* WrP  = (bf16_t*)(ws + off); off = align256(off + (size_t)HID * G4 * 2);
    bf16_t* WfcP = (bf16_t*)(ws + off); off = align256(off + (size_t)HID * VOCAB * 2);
    bf16_t* eP   = (bf16_t*)(ws + off); off = align256(off + (size_t)N_BATCH * EMB * 2);
    bf16_t* hA   = (bf16_t*)(ws + off); off = align256(off + (size_t)N_BATCH * HID * 2);
    float*  c    = (float*)(ws + off);  off = align256(off + (size_t)N_BATCH * HID * 4);
    float*  gates= (float*)(ws + off);  off = align256(off + (size_t)N_BATCH * G4 * 4);
    float*  logits=(float*)(ws + off);  off = align256(off + (size_t)N_BATCH * VOCAB * 4);
    int*    tok  = (int*)(ws + off);    off = align256(off + (size_t)N_BATCH * 4);

    float* probs_out = (float*)d_out;                                     // (N,T,V)
    float* tok_out   = (float*)d_out + (size_t)N_BATCH * T_STEPS * VOCAB; // (N,T)

    // Per-launch weight packing (deterministic; recomputed every call).
    {
        long long e;
        e = (long long)EMB * G4;
        pack_bf16_kernel<<<(unsigned)((e + 255) / 256), 256, 0, stream>>>(Wk, WkP, EMB, G4);
        e = (long long)HID * G4;
        pack_bf16_kernel<<<(unsigned)((e + 255) / 256), 256, 0, stream>>>(Wr, WrP, HID, G4);
        e = (long long)HID * VOCAB;
        pack_bf16_kernel<<<(unsigned)((e + 255) / 256), 256, 0, stream>>>(Wfc, WfcP, HID, VOCAB);
    }

    init_kernel<<<(N_BATCH * HID + 255) / 256, 256, 0, stream>>>(c0, c, tok);
    pack_act_kernel<<<(N_BATCH * HID + 255) / 256, 256, 0, stream>>>(h0, hA, HID);

    for (int t = 0; t < T_STEPS; ++t) {
        pack_emb_kernel<<<(N_BATCH * EMB + 255) / 256, 256, 0, stream>>>(emb, tok, eP);
        gates_kernel<<<G4 / 16 / 8, 128, 0, stream>>>(eP, WkP, hA, WrP, b_lstm, gates);
        lstm_kernel<<<(N_BATCH * HID + 255) / 256, 256, 0, stream>>>(gates, c, hA);
        logits_kernel<<<VOCAB / 16 / 8, 128, 0, stream>>>(hA, WfcP, bfc, logits);
        softmax_kernel<<<N_BATCH, 256, 0, stream>>>(logits, probs_out, tok_out, tok, t);
    }
}
